// ReLlamaAttention_60790967108168
// MI455X (gfx1250) — matrix operand order repaired and checked by reading, not compiled, not run on hardware
//
#include <hip/hip_runtime.h>
#include <math.h>

typedef __bf16 bf16_t;
typedef __attribute__((ext_vector_type(16))) __bf16 v16bf;
typedef __attribute__((ext_vector_type(8)))  float  v8f;

union Frag16 {
  v16bf v;
  bf16_t h[16];
  uint4  u[2];
};

// ---- problem constants ----
constexpr int S_    = 2048;
constexpr int H_    = 2048;
constexpr int NH_   = 32;
constexpr int NKV_  = 8;
constexpr int HD_   = 64;
constexpr int RBS_  = 64;
constexpr int ICS_  = 64;
constexpr int RBN_  = 16;
constexpr int RLEN_ = 1024;           // RBS*RBN
constexpr int KV_   = RLEN_ + S_;     // 3072
constexpr int START_= RLEN_;
constexpr float SCALE_ = 0.125f;      // 64^-0.5
constexpr float NEG_   = -1.0e9f;

__device__ __forceinline__ bf16_t f2bf(float x) {
  unsigned u = __builtin_bit_cast(unsigned, x);
  unsigned r = u + 0x7fffu + ((u >> 16) & 1u);
  unsigned short hs = (unsigned short)(r >> 16);
  return __builtin_bit_cast(bf16_t, hs);
}

__device__ __forceinline__ int kmapA(int i, int hl) {
  // 16-bit A-matrix 16x32 per-lane element->K mapping (ISA 7.12.2)
  return (i < 8) ? (hl * 8 + i) : (16 + hl * 8 + (i - 8));
}

__device__ __forceinline__ v8f wmma_bf16(v16bf a, v16bf b, v8f c) {
  return __builtin_amdgcn_wmma_f32_16x16x32_bf16(false, a, false, b, (short)0, c,
                                                 false, false);
}

// async global->LDS 16B transfer (per lane), tracked by ASYNCcnt
__device__ __forceinline__ void async_b128(unsigned lds_off, const void* gaddr) {
  asm volatile("global_load_async_to_lds_b128 %0, %1, off"
               :: "v"(lds_off), "v"(gaddr) : "memory");
}
__device__ __forceinline__ unsigned lds_addr(const void* p) {
  // LDS aperture: flat addr[31:0] is the LDS byte address (ISA 10.2)
  return (unsigned)(unsigned long long)p;
}

__device__ __forceinline__ bool mask_ok(int qpos, int kv) {
  bool causal = (qpos >= kv) && (kv > START_);
  int idk;
  if (kv < START_) {
    idk = kv / RBS_ + 1;
  } else {
    int ch = (kv - START_) / ICS_;
    idk = (ch == 0) ? -1 : (ch < RBN_ ? ch : RBN_);
  }
  int chq = (qpos - START_) / ICS_;
  int idq = (chq == 0) ? -1 : (chq < RBN_ ? chq : RBN_);
  bool doc = (idq == idk) && (kv < START_);
  return causal || doc;
}

// ---------------- elementwise kernels ----------------

__global__ void cast_bf16_kernel(const float* __restrict__ in, bf16_t* __restrict__ o, int n) {
  int i = blockIdx.x * blockDim.x + threadIdx.x;
  if (i < n) o[i] = f2bf(in[i]);
}

// in: [K][N] f32 row-major -> out: [N][K] bf16 (transposed)
__global__ void cast_transpose_bf16_kernel(const float* __restrict__ in,
                                           bf16_t* __restrict__ o, int K, int N) {
  int idx = blockIdx.x * blockDim.x + threadIdx.x;
  if (idx >= K * N) return;
  int n = idx % N;
  int k = idx / N;
  o[(size_t)n * K + k] = f2bf(in[idx]);
}

__global__ void rope_q_kernel(const float* __restrict__ q, const float* __restrict__ cb,
                              const float* __restrict__ sb, bf16_t* __restrict__ qb, int n) {
  int idx = blockIdx.x * blockDim.x + threadIdx.x;
  if (idx >= n) return;
  int d = idx % HD_;
  int s = idx / (NH_ * HD_);
  float x = q[idx];
  float rot = (d < HD_ / 2) ? -q[idx + HD_ / 2] : q[idx - HD_ / 2];
  float c = cb[s * HD_ + d], sn = sb[s * HD_ + d];
  qb[idx] = f2bf(x * c + rot * sn);
}

// k cache: [NKV][KV][HD]; v cache TRANSPOSED: [NKV][HD][KV]
__global__ void rope_kv_kernel(const float* __restrict__ k, const float* __restrict__ v,
                               const float* __restrict__ cb, const float* __restrict__ sb,
                               bf16_t* __restrict__ kc, bf16_t* __restrict__ vt, int n) {
  int idx = blockIdx.x * blockDim.x + threadIdx.x;
  if (idx >= n) return;
  int d  = idx % HD_;
  int kh = (idx / HD_) % NKV_;
  int s  = idx / (NKV_ * HD_);
  float x = k[idx];
  float rot = (d < HD_ / 2) ? -k[idx + HD_ / 2] : k[idx - HD_ / 2];
  float c = cb[s * HD_ + d], sn = sb[s * HD_ + d];
  kc[((size_t)kh * KV_ + START_ + s) * HD_ + d] = f2bf(x * c + rot * sn);
  vt[((size_t)kh * HD_ + d) * KV_ + START_ + s] = f2bf(v[idx]);
}

__global__ void pack_ret_kernel(const float* __restrict__ rk, const float* __restrict__ rv,
                                bf16_t* __restrict__ kc, bf16_t* __restrict__ vt, int n) {
  int idx = blockIdx.x * blockDim.x + threadIdx.x;
  if (idx >= n) return;
  int d   = idx % HD_;
  int row = (idx / HD_) % RLEN_;
  int kh  = idx / (RLEN_ * HD_);
  kc[((size_t)kh * KV_ + row) * HD_ + d] = f2bf(rk[idx]);
  vt[((size_t)kh * HD_ + d) * KV_ + row] = f2bf(rv[idx]);
}

// ---------------- WMMA GEMM: C[M,N] = A[M,K] * Bt[N,K]^T ----------------
// Double-buffered async global->LDS staging (ASYNCcnt), 64x64 block tile,
// K-step 32, 8 waves x 2 WMMA per k-step, all fragment gathers ds_load_b128.

__global__ __launch_bounds__(256) void gemm_bf16_wmma(const bf16_t* __restrict__ A,
                                                      const bf16_t* __restrict__ Bt,
                                                      float* __restrict__ C,
                                                      int M, int N, int K) {
  __shared__ bf16_t As[2][64][32 + 8];   // pitch 40 elems = 80B
  __shared__ bf16_t Bs[2][64][32 + 8];   // [n][k]

  int tid  = threadIdx.x;
  int lane = tid & 31;
  int wave = tid >> 5;      // 0..7
  int wm   = wave >> 1;     // 0..3  -> M tile (16 rows)
  int wn   = wave & 1;      // 0..1  -> N strip (32 cols)
  int m0 = blockIdx.y * 64;
  int n0 = blockIdx.x * 64;

  int hl = lane >> 4;       // half (0/1)
  int ml = lane & 15;

  // cooperative staging: 64 rows x 32 cols bf16, one 16B async op / thread / buf
  int lr = tid >> 2;            // 0..63
  int lc = (tid & 3) * 8;       // 0,8,16,24

  const bf16_t* ga = A  + (size_t)(m0 + lr) * K + lc;
  const bf16_t* gb = Bt + (size_t)(n0 + lr) * K + lc;
  unsigned la0 = lds_addr(&As[0][lr][lc]), la1 = lds_addr(&As[1][lr][lc]);
  unsigned lb0 = lds_addr(&Bs[0][lr][lc]), lb1 = lds_addr(&Bs[1][lr][lc]);

  v8f acc0 = {}, acc1 = {};

  int nk = K >> 5;
  // prologue: tile 0 -> buffer 0
  async_b128(la0, ga);
  async_b128(lb0, gb);

  for (int t = 0; t < nk; ++t) {
    int cur = t & 1;
    if (t + 1 < nk) {
      // issue next tile into the other buffer (2 more ASYNCcnt in flight)
      async_b128(cur ? la0 : la1, ga + (t + 1) * 32);
      async_b128(cur ? lb0 : lb1, gb + (t + 1) * 32);
      asm volatile("s_wait_asynccnt 0x2" ::: "memory");   // tile t resident
    } else {
      asm volatile("s_wait_asynccnt 0x0" ::: "memory");
    }
    if (t + 2 < nk) {
      __builtin_prefetch(ga + (t + 2) * 32, 0, 1);
      __builtin_prefetch(gb + (t + 2) * 32, 0, 1);
    }
    __syncthreads();

    Frag16 af, bf0, bf1;
    {
      int arow = wm * 16 + ml;
      af.u[0] = *(const uint4*)&As[cur][arow][hl * 8];
      af.u[1] = *(const uint4*)&As[cur][arow][16 + hl * 8];
    }
    {
      int bn = wn * 32 + ml;                 // column of C
      bf0.u[0] = *(const uint4*)&Bs[cur][bn][hl * 8];
      bf0.u[1] = *(const uint4*)&Bs[cur][bn][16 + hl * 8];
      bf1.u[0] = *(const uint4*)&Bs[cur][bn + 16][hl * 8];
      bf1.u[1] = *(const uint4*)&Bs[cur][bn + 16][16 + hl * 8];
    }
    acc0 = wmma_bf16(af.v, bf0.v, acc0);
    acc1 = wmma_bf16(af.v, bf1.v, acc1);
    __syncthreads();   // all waves done reading buffer `cur` before tile t+2 lands
  }

  int rowb = m0 + wm * 16 + hl * 8;
  int col  = n0 + wn * 32 + ml;
#pragma unroll
  for (int r = 0; r < 8; ++r) {
    C[(size_t)(rowb + r) * N + col]      = acc0[r];
    C[(size_t)(rowb + r) * N + col + 16] = acc1[r];
  }
}

// ---------------- Flash attention ----------------
// grid (S/64, NH), block 128 (4 waves); each wave owns 16 q rows of one head.
// K cache [NKV][KV][HD] (d-contiguous), V cache transposed [NKV][HD][KV]
// (kv-contiguous). Double-buffered async K/V tile staging.

__global__ __launch_bounds__(128) void attn_kernel(const bf16_t* __restrict__ Qb,   // [S][NH*HD]
                                                   const bf16_t* __restrict__ Kc,   // [NKV][KV][HD]
                                                   const bf16_t* __restrict__ Vt,   // [NKV][HD][KV]
                                                   float* __restrict__ Out) {       // [S][NH*HD]
  __shared__ bf16_t Ks[2][32][HD_ + 8];     // [kv][d], pitch 72
  __shared__ bf16_t Vs[2][HD_][32 + 8];     // [d][kv], pitch 40
  __shared__ bf16_t Ps[4][16][32 + 8];      // per-wave P staging, pitch 40

  int tid  = threadIdx.x;
  int lane = tid & 31;
  int wave = tid >> 5;
  int hl   = lane >> 4;
  int ml   = lane & 15;

  int h  = blockIdx.y;
  int hk = h >> 2;                          // NH/NKV = 4
  int qbase = blockIdx.x * 64 + wave * 16;

  // preload Q A-fragments (two HD slices of 32)
  Frag16 qa[2];
  {
    int qrow = qbase + ml;
#pragma unroll
    for (int sl = 0; sl < 2; ++sl)
#pragma unroll
      for (int i = 0; i < 16; ++i)
        qa[sl].h[i] = Qb[(size_t)qrow * (NH_ * HD_) + h * HD_ + sl * 32 + kmapA(i, hl)];
  }

  v8f o0 = {}, o1 = {}, o2 = {}, o3 = {};
  float mrun[8], lrun[8];
#pragma unroll
  for (int r = 0; r < 8; ++r) { mrun[r] = NEG_; lrun[r] = 0.f; }

  // K tile staging: 32 rows x 64 cols, 2x16B async / thread
  int kr_ = tid >> 2;
  int kc_ = (tid & 3) * 16;
  // V tile staging: 64 rows x 32 cols, 2x16B async / thread
  int vr_ = tid >> 1;
  int vc_ = (tid & 1) * 16;

  const bf16_t* gk = Kc + ((size_t)hk * KV_ + kr_) * HD_ + kc_;
  const bf16_t* gv = Vt + ((size_t)hk * HD_ + vr_) * KV_ + vc_;
  unsigned lk0a = lds_addr(&Ks[0][kr_][kc_]),     lk1a = lds_addr(&Ks[1][kr_][kc_]);
  unsigned lk0b = lds_addr(&Ks[0][kr_][kc_ + 8]), lk1b = lds_addr(&Ks[1][kr_][kc_ + 8]);
  unsigned lv0a = lds_addr(&Vs[0][vr_][vc_]),     lv1a = lds_addr(&Vs[1][vr_][vc_]);
  unsigned lv0b = lds_addr(&Vs[0][vr_][vc_ + 8]), lv1b = lds_addr(&Vs[1][vr_][vc_ + 8]);

  const int NT = KV_ / 32;   // 96 tiles
  // prologue: tile 0 -> buffer 0 (4 async ops / thread / tile)
  async_b128(lk0a, gk);
  async_b128(lk0b, gk + 8);
  async_b128(lv0a, gv);
  async_b128(lv0b, gv + 8);

  for (int t = 0; t < NT; ++t) {
    int cur = t & 1;
    int kv0 = t * 32;
    if (t + 1 < NT) {
      const bf16_t* nk_ = gk + (size_t)(t + 1) * 32 * HD_;
      const bf16_t* nv_ = gv + (t + 1) * 32;
      async_b128(cur ? lk0a : lk1a, nk_);
      async_b128(cur ? lk0b : lk1b, nk_ + 8);
      async_b128(cur ? lv0a : lv1a, nv_);
      async_b128(cur ? lv0b : lv1b, nv_ + 8);
      asm volatile("s_wait_asynccnt 0x4" ::: "memory");   // tile t resident
    } else {
      asm volatile("s_wait_asynccnt 0x0" ::: "memory");
    }
    __syncthreads();

    // ---- scores: two 16x16 tiles, each 2 WMMAs over HD ----
    v8f sc0 = {}, sc1 = {};
#pragma unroll
    for (int sl = 0; sl < 2; ++sl) {
      Frag16 kb0, kb1;
      const uint4* p0 = (const uint4*)&Ks[cur][ml][sl * 32 + hl * 8];
      const uint4* p1 = (const uint4*)&Ks[cur][16 + ml][sl * 32 + hl * 8];
      kb0.u[0] = p0[0]; kb0.u[1] = p0[2];
      kb1.u[0] = p1[0]; kb1.u[1] = p1[2];
      sc0 = wmma_bf16(qa[sl].v, kb0.v, sc0);
      sc1 = wmma_bf16(qa[sl].v, kb1.v, sc1);
    }

    // ---- scale + mask + online softmax ----
    float p0a[8], p1a[8];
#pragma unroll
    for (int r = 0; r < 8; ++r) {
      int qpos = qbase + r + 8 * hl + START_;
      int kva = kv0 + ml;
      int kvb = kv0 + 16 + ml;
      float va = sc0[r] * SCALE_;
      float vb = sc1[r] * SCALE_;
      va = mask_ok(qpos, kva) ? va : NEG_;
      vb = mask_ok(qpos, kvb) ? vb : NEG_;
      float mx = fmaxf(va, vb);
#pragma unroll
      for (int m = 1; m < 16; m <<= 1) mx = fmaxf(mx, __shfl_xor(mx, m, 32));
      float mnew = fmaxf(mrun[r], mx);
      float alpha = __expf(mrun[r] - mnew);
      mrun[r] = mnew;
      float pa = (va <= -1.0e8f) ? 0.f : __expf(va - mnew);
      float pb = (vb <= -1.0e8f) ? 0.f : __expf(vb - mnew);
      float psum = pa + pb;
#pragma unroll
      for (int m = 1; m < 16; m <<= 1) psum += __shfl_xor(psum, m, 32);
      lrun[r] = lrun[r] * alpha + psum;
      o0[r] *= alpha; o1[r] *= alpha; o2[r] *= alpha; o3[r] *= alpha;
      p0a[r] = pa; p1a[r] = pb;
    }

    // ---- stage P (C-layout -> A-layout) through wave-private LDS ----
#pragma unroll
    for (int r = 0; r < 8; ++r) {
      int row = r + 8 * hl;
      Ps[wave][row][ml]      = f2bf(p0a[r]);
      Ps[wave][row][16 + ml] = f2bf(p1a[r]);
    }
    asm volatile("s_wait_dscnt 0" ::: "memory");

    Frag16 pa;
    pa.u[0] = *(const uint4*)&Ps[wave][ml][hl * 16];
    pa.u[1] = *(const uint4*)&Ps[wave][ml][hl * 16 + 8];

    // ---- P @ V : 4 WMMAs over HD, V fragments contiguous from Vs[d][kv] ----
    Frag16 vb0, vb1, vb2, vb3;
    vb0.u[0] = *(const uint4*)&Vs[cur][ml][hl * 16];
    vb0.u[1] = *(const uint4*)&Vs[cur][ml][hl * 16 + 8];
    vb1.u[0] = *(const uint4*)&Vs[cur][16 + ml][hl * 16];
    vb1.u[1] = *(const uint4*)&Vs[cur][16 + ml][hl * 16 + 8];
    vb2.u[0] = *(const uint4*)&Vs[cur][32 + ml][hl * 16];
    vb2.u[1] = *(const uint4*)&Vs[cur][32 + ml][hl * 16 + 8];
    vb3.u[0] = *(const uint4*)&Vs[cur][48 + ml][hl * 16];
    vb3.u[1] = *(const uint4*)&Vs[cur][48 + ml][hl * 16 + 8];
    o0 = wmma_bf16(pa.v, vb0.v, o0);
    o1 = wmma_bf16(pa.v, vb1.v, o1);
    o2 = wmma_bf16(pa.v, vb2.v, o2);
    o3 = wmma_bf16(pa.v, vb3.v, o3);

    __syncthreads();   // all waves done with buffer `cur` before tile t+2 lands
  }

  // ---- finalize: divide by row sum, zero invalid rows ----
#pragma unroll
  for (int r = 0; r < 8; ++r) {
    float inv = (lrun[r] > 0.f) ? 1.0f / lrun[r] : 0.0f;
    int srow = qbase + r + 8 * hl;
    size_t base = (size_t)srow * (NH_ * HD_) + h * HD_;
    Out[base + ml]      = o0[r] * inv;
    Out[base + 16 + ml] = o1[r] * inv;
    Out[base + 32 + ml] = o2[r] * inv;
    Out[base + 48 + ml] = o3[r] * inv;
  }
}

// ---------------- host launcher ----------------

extern "C" void kernel_launch(void* const* d_in, const int* in_sizes, int n_in,
                              void* d_out, int out_size, void* d_ws, size_t ws_size,
                              hipStream_t stream) {
  const float* hidden = (const float*)d_in[0];
  const float* cosb   = (const float*)d_in[1];
  const float* sinb   = (const float*)d_in[2];
  const float* retK   = (const float*)d_in[3];
  const float* retV   = (const float*)d_in[4];
  const float* Wq     = (const float*)d_in[5];
  const float* Wk     = (const float*)d_in[6];
  const float* Wv     = (const float*)d_in[7];
  const float* Wo     = (const float*)d_in[8];
  float* out = (float*)d_out;
  (void)in_sizes; (void)n_in; (void)out_size; (void)ws_size;

  char* ws = (char*)d_ws;
  size_t off = 0;
  auto take = [&](size_t bytes) -> char* {
    char* p = ws + off;
    off += (bytes + 255) & ~(size_t)255;
    return p;
  };

  bf16_t* hb  = (bf16_t*)take((size_t)S_ * H_ * 2);              // hidden bf16
  bf16_t* wqt = (bf16_t*)take((size_t)H_ * NH_ * HD_ * 2);       // Wq^T bf16 [N][K]
  bf16_t* wkt = (bf16_t*)take((size_t)H_ * NKV_ * HD_ * 2);      // Wk^T bf16
  bf16_t* wvt = (bf16_t*)take((size_t)H_ * NKV_ * HD_ * 2);      // Wv^T bf16
  bf16_t* wot = (bf16_t*)take((size_t)NH_ * HD_ * H_ * 2);       // Wo^T bf16
  float*  qf  = (float*)take((size_t)S_ * NH_ * HD_ * 4);        // q proj f32
  float*  kf  = (float*)take((size_t)S_ * NKV_ * HD_ * 4);       // k proj f32
  float*  vf  = (float*)take((size_t)S_ * NKV_ * HD_ * 4);       // v proj f32
  bf16_t* qb  = (bf16_t*)take((size_t)S_ * NH_ * HD_ * 2);       // roped q bf16
  bf16_t* kc  = (bf16_t*)take((size_t)NKV_ * KV_ * HD_ * 2);     // K cache [NKV][KV][HD]
  bf16_t* vt  = (bf16_t*)take((size_t)NKV_ * HD_ * KV_ * 2);     // V cache [NKV][HD][KV]
  float*  ao  = (float*)take((size_t)S_ * NH_ * HD_ * 4);        // attn out f32
  bf16_t* aob = (bf16_t*)take((size_t)S_ * NH_ * HD_ * 2);       // attn out bf16

  const int T = 256;
  auto blocks = [](int n, int t) { return (n + t - 1) / t; };

  // 1) casts (weights transposed for K-contiguous B fragments)
  int nh_ = S_ * H_;
  cast_bf16_kernel<<<blocks(nh_, T), T, 0, stream>>>(hidden, hb, nh_);
  int nq = H_ * NH_ * HD_;
  cast_transpose_bf16_kernel<<<blocks(nq, T), T, 0, stream>>>(Wq, wqt, H_, NH_ * HD_);
  int nk = H_ * NKV_ * HD_;
  cast_transpose_bf16_kernel<<<blocks(nk, T), T, 0, stream>>>(Wk, wkt, H_, NKV_ * HD_);
  cast_transpose_bf16_kernel<<<blocks(nk, T), T, 0, stream>>>(Wv, wvt, H_, NKV_ * HD_);
  int no = NH_ * HD_ * H_;
  cast_transpose_bf16_kernel<<<blocks(no, T), T, 0, stream>>>(Wo, wot, NH_ * HD_, H_);

  // 2) QKV projections (WMMA GEMM, async double-buffered tiles)
  gemm_bf16_wmma<<<dim3((NH_ * HD_) / 64, S_ / 64), 256, 0, stream>>>(hb, wqt, qf, S_, NH_ * HD_, H_);
  gemm_bf16_wmma<<<dim3((NKV_ * HD_) / 64, S_ / 64), 256, 0, stream>>>(hb, wkt, kf, S_, NKV_ * HD_, H_);
  gemm_bf16_wmma<<<dim3((NKV_ * HD_) / 64, S_ / 64), 256, 0, stream>>>(hb, wvt, vf, S_, NKV_ * HD_, H_);

  // 3) RoPE + cache packing (V packed transposed)
  int nqe = S_ * NH_ * HD_;
  rope_q_kernel<<<blocks(nqe, T), T, 0, stream>>>(qf, cosb, sinb, qb, nqe);
  int nke = S_ * NKV_ * HD_;
  rope_kv_kernel<<<blocks(nke, T), T, 0, stream>>>(kf, vf, cosb, sinb, kc, vt, nke);
  int nre = NKV_ * RLEN_ * HD_;
  pack_ret_kernel<<<blocks(nre, T), T, 0, stream>>>(retK, retV, kc, vt, nre);

  // 4) flash attention (WMMA, async double-buffered K/V tiles)
  attn_kernel<<<dim3(S_ / 64, NH_), 128, 0, stream>>>(qb, kc, vt, ao);

  // 5) output projection (WMMA GEMM)
  cast_bf16_kernel<<<blocks(nqe, T), T, 0, stream>>>(ao, aob, nqe);
  gemm_bf16_wmma<<<dim3(H_ / 64, S_ / 64), 256, 0, stream>>>(aob, wot, out, S_, H_, NH_ * HD_);
}
